// GraphConstructionActorHead_40724879901154
// MI455X (gfx1250) — matrix operand order repaired and checked by reading, not compiled, not run on hardware
//
#include <hip/hip_runtime.h>

typedef __attribute__((ext_vector_type(16))) _Float16 v16h;
typedef __attribute__((ext_vector_type(8)))  float    v8f;

#define NN 1024
#define DD 64
#define HH 64
#define TILES ((NN * NN) / 16)
#define LN_EPS 1e-5f

// ---------------------------------------------------------------------------
// Phase 1: A[i][h] = emb[i] @ W1[:64] + b1 ;  B[i][h] = emb[i] @ W1[64:]
// Tiny (1024x64 each) -> plain FMA kernel; result lives in L2 for phase 2.
// ---------------------------------------------------------------------------
__global__ void precompute_ab(const float* __restrict__ emb,
                              const float* __restrict__ W1,
                              const float* __restrict__ b1,
                              float* __restrict__ A,
                              float* __restrict__ B) {
  const int i = blockIdx.x;
  const int h = threadIdx.x;           // 64 threads
  const float* e = emb + i * DD;
  float a = b1[h];
  float b = 0.0f;
#pragma unroll 8
  for (int d = 0; d < DD; ++d) {
    const float ev = e[d];
    a = fmaf(ev, W1[d * HH + h], a);
    b = fmaf(ev, W1[(DD + d) * HH + h], b);
  }
  A[i * HH + h] = a;
  B[i * HH + h] = b;
}

// ---------------------------------------------------------------------------
// Phase 2: per 16-pair tile (one wave):
//   h0 = A[i] + B[j]            (layer-1 result, factorized)
//   h1 = relu(LN(h0; g1,be1))   -> packed into f16 WMMA A-fragments
//   h2 = h1 @ W2 + b2           -> v_wmma_f32_16x16x32_f16 x 8
//   score = relu(LN(h2; g2,be2)) @ W3 + b3
//   out = score * valid_mask * (i != j)
// ---------------------------------------------------------------------------
__global__ void __launch_bounds__(256)
pair_mlp(const float* __restrict__ A, const float* __restrict__ B,
         const float* __restrict__ g1, const float* __restrict__ be1,
         const float* __restrict__ W2, const float* __restrict__ b2,
         const float* __restrict__ g2, const float* __restrict__ be2,
         const float* __restrict__ W3, const float* __restrict__ b3,
         const float* __restrict__ mask, float* __restrict__ out) {
  const int lane  = threadIdx.x & 31;
  const int l16   = lane & 15;
  const int half  = lane >> 4;         // 0: K-subset base 0, 1: base 8
  const int kbase = half * 8;
  const int wave   = blockIdx.x * (blockDim.x >> 5) + (threadIdx.x >> 5);
  const int nwaves = gridDim.x * (blockDim.x >> 5);

  // ---- persistent W2 B-fragments: bfr[ntile][kchunk] --------------------
  // Dense 16-bit B (32x16): lane<16 -> col=lane, K=0..7,16..23 ; lane>=16 -> col=lane-16, K=8..15,24..31
  v16h bfr[4][2];
#pragma unroll
  for (int nt = 0; nt < 4; ++nt) {
#pragma unroll
    for (int kc = 0; kc < 2; ++kc) {
      const int col = nt * 16 + l16;
      const int k0  = kc * 32 + half * 8;
      v16h f;
#pragma unroll
      for (int e = 0; e < 16; ++e)
        f[e] = (_Float16)W2[(k0 + ((e < 8) ? e : (e + 8))) * HH + col];
      bfr[nt][kc] = f;
    }
  }

  // ---- per-lane LN1 params for the K positions this lane owns (f16-packed)
  // A-fragment element convention: elem e (chunk c) <-> K = 32c + kbase + (e<8 ? e : e+8)
  v16h g1f[2], be1f[2];
#pragma unroll
  for (int c = 0; c < 2; ++c) {
    v16h gt, bt;
#pragma unroll
    for (int e = 0; e < 16; ++e) {
      const int k = c * 32 + kbase + ((e < 8) ? e : (e + 8));
      gt[e] = (_Float16)g1[k];
      bt[e] = (_Float16)be1[k];
    }
    g1f[c] = gt;
    be1f[c] = bt;
  }

  // ---- per-lane LN2 / head params for columns n = 16*nt + l16 -----------
  float b2v[4], g2v[4], be2v[4], w3v[4];
#pragma unroll
  for (int nt = 0; nt < 4; ++nt) {
    const int n = nt * 16 + l16;
    b2v[nt]  = b2[n];
    g2v[nt]  = g2[n];
    be2v[nt] = be2[n];
    w3v[nt]  = W3[n];
  }
  const float b3s = b3[0];

  for (int t = wave; t < TILES; t += nwaves) {
    const int i  = t >> 6;             // source node (fixed per tile)
    const int j0 = (t & 63) << 4;      // 16 consecutive target nodes
    const int j  = j0 + l16;           // this lane's pair row (lanes L and L+16 share row L)

    // ---- layer 1 (factorized) + LN1 stats ------------------------------
    const float4* Ai4 = (const float4*)(A + i * HH);
    const float4* Bj4 = (const float4*)(B + j * HH);
    float h0v[2][16];
    float s = 0.0f, sq = 0.0f;
#pragma unroll
    for (int c = 0; c < 2; ++c) {
#pragma unroll
      for (int q = 0; q < 4; ++q) {    // elems 4q..4q+3; contiguous K quad
        const int k  = c * 32 + kbase + ((q < 2) ? 4 * q : 4 * q + 8);
        const float4 av = Ai4[k >> 2];
        const float4 bv = Bj4[k >> 2];
        const float v0 = av.x + bv.x, v1 = av.y + bv.y;
        const float v2 = av.z + bv.z, v3 = av.w + bv.w;
        h0v[c][4 * q + 0] = v0; h0v[c][4 * q + 1] = v1;
        h0v[c][4 * q + 2] = v2; h0v[c][4 * q + 3] = v3;
        s  += v0 + v1 + v2 + v3;
        sq += v0 * v0 + v1 * v1 + v2 * v2 + v3 * v3;
      }
    }
    // lanes L and L+16 jointly own row L: one xor-16 exchange completes the row
    s  += __shfl_xor(s, 16, 32);
    sq += __shfl_xor(sq, 16, 32);
    const float mean = s * (1.0f / 64.0f);
    const float rstd = rsqrtf(sq * (1.0f / 64.0f) - mean * mean + LN_EPS);

    // ---- LN1 + ReLU, pack f16 A-fragments ------------------------------
    v16h afr[2];
#pragma unroll
    for (int c = 0; c < 2; ++c) {
      v16h a;
#pragma unroll
      for (int e = 0; e < 16; ++e) {
        float v = (h0v[c][e] - mean) * rstd * (float)g1f[c][e] + (float)be1f[c][e];
        v = v > 0.0f ? v : 0.0f;
        a[e] = (_Float16)v;
      }
      afr[c] = a;
    }

    // ---- hidden GEMM: 16x64 = (16x64) @ (64x64) via 8 WMMAs ------------
    v8f cacc[4] = {};
#pragma unroll
    for (int nt = 0; nt < 4; ++nt) {
      cacc[nt] = __builtin_amdgcn_wmma_f32_16x16x32_f16(
          false, afr[0], false, bfr[nt][0], (short)0, cacc[nt], false, false);
      cacc[nt] = __builtin_amdgcn_wmma_f32_16x16x32_f16(
          false, afr[1], false, bfr[nt][1], (short)0, cacc[nt], false, false);
    }

    // ---- + b2, LN2, ReLU, dot W3 ---------------------------------------
    // C layout: elem r <-> row m = half*8 + r, col n = 16*nt + l16
    float sc[8];
#pragma unroll
    for (int r = 0; r < 8; ++r) {
      float hv[4];
      float s2 = 0.0f, q2 = 0.0f;
#pragma unroll
      for (int nt = 0; nt < 4; ++nt) {
        const float v = cacc[nt][r] + b2v[nt];
        hv[nt] = v;
        s2 += v;
        q2 += v * v;
      }
      // reduce over the 16 lanes of this half (xor masks 1,2,4,8 stay in-half)
      s2 += __shfl_xor(s2, 1, 32); s2 += __shfl_xor(s2, 2, 32);
      s2 += __shfl_xor(s2, 4, 32); s2 += __shfl_xor(s2, 8, 32);
      q2 += __shfl_xor(q2, 1, 32); q2 += __shfl_xor(q2, 2, 32);
      q2 += __shfl_xor(q2, 4, 32); q2 += __shfl_xor(q2, 8, 32);
      const float m2 = s2 * (1.0f / 64.0f);
      const float r2 = rsqrtf(q2 * (1.0f / 64.0f) - m2 * m2 + LN_EPS);
      float acc = 0.0f;
#pragma unroll
      for (int nt = 0; nt < 4; ++nt) {
        float v = (hv[nt] - m2) * r2 * g2v[nt] + be2v[nt];
        v = v > 0.0f ? v : 0.0f;
        acc = fmaf(v, w3v[nt], acc);
      }
      acc += __shfl_xor(acc, 1, 32); acc += __shfl_xor(acc, 2, 32);
      acc += __shfl_xor(acc, 4, 32); acc += __shfl_xor(acc, 8, 32);
      sc[r] = acc + b3s;
    }

    // ---- masked store: lanes l16==r write rows half*8 + r ---------------
#pragma unroll
    for (int r = 0; r < 8; ++r) {
      if (l16 == r) {
        const int m  = half * 8 + r;
        const int jj = j0 + m;
        const int p  = i * NN + jj;
        out[p] = sc[r] * mask[p] * ((jj == i) ? 0.0f : 1.0f);
      }
    }
  }
}

extern "C" void kernel_launch(void* const* d_in, const int* in_sizes, int n_in,
                              void* d_out, int out_size, void* d_ws, size_t ws_size,
                              hipStream_t stream) {
  const float* emb  = (const float*)d_in[0];
  const float* mask = (const float*)d_in[1];
  const float* W1   = (const float*)d_in[2];
  const float* b1   = (const float*)d_in[3];
  const float* g1   = (const float*)d_in[4];
  const float* be1  = (const float*)d_in[5];
  const float* W2   = (const float*)d_in[6];
  const float* b2   = (const float*)d_in[7];
  const float* g2   = (const float*)d_in[8];
  const float* be2  = (const float*)d_in[9];
  const float* W3   = (const float*)d_in[10];
  const float* b3   = (const float*)d_in[11];
  float* out = (float*)d_out;

  float* A = (float*)d_ws;             // 1024*64 floats
  float* B = A + NN * HH;              // 1024*64 floats (512 KB total)

  precompute_ab<<<NN, HH, 0, stream>>>(emb, W1, b1, A, B);
  pair_mlp<<<1024, 256, 0, stream>>>(A, B, g1, be1, W2, b2, g2, be2, W3, b3,
                                     mask, out);
}
